// EnVAE_21114059227421
// MI455X (gfx1250) — matrix operand order repaired and checked by reading, not compiled, not run on hardware
//
#include <hip/hip_runtime.h>
#include <hip/hip_bf16.h>

typedef __attribute__((ext_vector_type(16))) __bf16 v16bf;
typedef __attribute__((ext_vector_type(8)))  __bf16 v8bf;
typedef __attribute__((ext_vector_type(2)))  __bf16 bf16x2;
typedef __attribute__((ext_vector_type(8)))  float  v8f;

#define OBS   1024
#define GS    128      // group size (K of GEMM1)
#define HID   128      // hidden (N of GEMM1, K/N of GEMM2)
#define BT    128      // batch rows per block
#define STR   136      // bf16 row stride: 16B-aligned, 68 words -> 4-bank shift/row
#define OSTR  132      // f32 row stride for selection scratch

// LDS carve (bytes); each bf16 region = 128*136*2 = 34816
#define OFF_XG   0
#define OFF_W1T  34816
#define OFF_W2T  69632
#define OFF_H    104448
#define OFF_B1   139264
#define OFF_B2   (139264 + 512)
#define SMEM_BYTES (139264 + 1024)
// sOut (f32, 128*132*4 = 67584 B) aliases sXg+sW1T (69632 B) after a barrier
#define OFF_OUT  0

// A fragment, 16x32 bf16 (ISA 7.12.2): lanes 0-15 row m=lane, K {0-7,16-23};
// lanes 16-31 row m=lane-16, K {8-15,24-31}. VGPR0-3 = first chunk, 4-7 = second.
__device__ __forceinline__ v16bf load_frag_a(const __bf16* base, int k0, int lane) {
  const int half = lane >> 4, m = lane & 15;
  const __bf16* p = base + m * STR + k0 + half * 8;
  union { v16bf v; v8bf h[2]; } u;
  u.h[0] = *(const v8bf*)(p);
  u.h[1] = *(const v8bf*)(p + 16);
  return u.v;
}

// B fragment, 32x16 bf16: lane = column n (mod 16); lanes 0-15 hold K {0-7,16-23},
// lanes 16-31 hold K {8-15,24-31}. Needs N-major (transposed) storage in LDS.
__device__ __forceinline__ v16bf load_frag_b(const __bf16* base, int k0, int lane) {
  const int half = lane >> 4, n = lane & 15;
  const __bf16* p = base + n * STR + k0 + half * 8;
  union { v16bf v; v8bf h[2]; } u;
  u.h[0] = *(const v8bf*)(p);
  u.h[1] = *(const v8bf*)(p + 16);
  return u.v;
}

__device__ __forceinline__ v8f bwmma(v16bf a, v16bf b, v8f c) {
  return __builtin_amdgcn_wmma_f32_16x16x32_bf16(false, a, false, b,
                                                 (short)0, c, false, false);
}

__device__ __forceinline__ unsigned pack2(float a, float b) {
  union { bf16x2 v; unsigned u; } t;
  t.v.x = (__bf16)a; t.v.y = (__bf16)b;
  return t.u;
}

// One wave computes 2 row-tiles x 4 col-tiles; B fragments shared by 2 WMMAs.
__device__ __forceinline__ void gemm_pair(const __bf16* aBase0, const __bf16* aBase1,
                                          const __bf16* bBase, int lane, v8f acc[8]) {
  #pragma unroll
  for (int ks = 0; ks < 4; ++ks) {
    const int k0 = ks * 32;
    v16bf a0 = load_frag_a(aBase0, k0, lane);
    v16bf a1 = load_frag_a(aBase1, k0, lane);
    #pragma unroll
    for (int c = 0; c < 4; ++c) {
      v16bf b = load_frag_b(bBase + c * 16 * STR, k0, lane);
      acc[c]     = bwmma(a0, b, acc[c]);
      acc[4 + c] = bwmma(a1, b, acc[4 + c]);
    }
  }
}

__global__ __launch_bounds__(256) void envae_kernel(
    const float* __restrict__ X, const float* __restrict__ eps,
    const float* __restrict__ W1, const float* __restrict__ b1,
    const float* __restrict__ W2, const float* __restrict__ b2,
    const int* __restrict__ indices, float* __restrict__ out, int batch)
{
  __shared__ __align__(16) char smem[SMEM_BYTES];
  __bf16* sXg  = (__bf16*)(smem + OFF_XG);
  __bf16* sW1T = (__bf16*)(smem + OFF_W1T);
  __bf16* sW2T = (__bf16*)(smem + OFF_W2T);
  __bf16* sH   = (__bf16*)(smem + OFF_H);
  float*  sB1  = (float*)(smem + OFF_B1);
  float*  sB2  = (float*)(smem + OFF_B2);
  float*  sOut = (float*)(smem + OFF_OUT);

  const int g    = blockIdx.x & 7;   // group fastest -> 8 blocks share one X tile in L2
  const int tile = blockIdx.x >> 3;
  const int tid  = threadIdx.x;
  const int lane = tid & 31;
  const int wid  = tid >> 5;

  const float* W1g = W1 + g * GS * HID;
  const float* W2g = W2 + g * HID * HID;
  const float* Xp  = X + (size_t)tile * BT * OBS + g;

  // ---- stage weights transposed, bf16x2-packed along i (both loads coalesced over o) ----
  for (int e = tid; e < (GS * HID) / 2; e += 256) {       // 32 iterations
    const int i = (e >> 7) * 2, o = e & 127;
    *(unsigned*)&sW1T[o * STR + i] = pack2(W1g[i * HID + o], W1g[(i + 1) * HID + o]);
    *(unsigned*)&sW2T[o * STR + i] = pack2(W2g[i * HID + o], W2g[(i + 1) * HID + o]);
  }
  if (tid < 128)      sB1[tid]       = b1[g * HID + tid];
  else                sB2[tid - 128] = b2[g * HID + (tid - 128)];
  // ---- stage Xg tile (stride-8 column gather of X), packed column pairs ----
  for (int e = tid; e < (BT * GS) / 2; e += 256) {        // 32 iterations
    const int r = e >> 6, c = (e & 63) * 2;
    const float x0 = Xp[(size_t)r * OBS + 8 * c];
    const float x1 = Xp[(size_t)r * OBS + 8 * c + 8];
    *(unsigned*)&sXg[r * STR + c] = pack2(x0, x1);
  }
  __syncthreads();

  // wave w: row-tiles {Rb, Rb+16}, col-tiles C0 + {0,16,32,48}
  const int Rb = 32 * (wid >> 1);
  const int C0 = (wid & 1) * 64;
  const int nlo = lane & 15, hb = lane >> 4;

  // ---- GEMM1: h = relu(Xg @ W1 + b1) ----
  v8f acc[8];
  #pragma unroll
  for (int n = 0; n < 8; ++n) { v8f z = {0.f,0.f,0.f,0.f,0.f,0.f,0.f,0.f}; acc[n] = z; }
  gemm_pair(sXg + Rb * STR, sXg + (Rb + 16) * STR, sW1T + C0 * STR, lane, acc);

  #pragma unroll
  for (int rt = 0; rt < 2; ++rt) {
    #pragma unroll
    for (int c = 0; c < 4; ++c) {
      const int col = C0 + c * 16 + nlo;
      const float bias = sB1[col];
      #pragma unroll
      for (int j = 0; j < 8; ++j) {
        float v = acc[rt * 4 + c][j] + bias;
        sH[(Rb + rt * 16 + hb * 8 + j) * STR + col] = (__bf16)(v > 0.f ? v : 0.f);
      }
    }
  }
  __syncthreads();   // sH rows are produced by wave pairs -> block-wide visibility

  // ---- GEMM2: out2 = h @ W2 + b2 ----
  v8f acc2[8];
  #pragma unroll
  for (int n = 0; n < 8; ++n) { v8f z = {0.f,0.f,0.f,0.f,0.f,0.f,0.f,0.f}; acc2[n] = z; }
  gemm_pair(sH + Rb * STR, sH + (Rb + 16) * STR, sW2T + C0 * STR, lane, acc2);

  __syncthreads();   // everyone done with sXg/sW1T; sOut may now alias them

  // spill out2 (f32) to LDS for per-row latent selection
  #pragma unroll
  for (int rt = 0; rt < 2; ++rt) {
    #pragma unroll
    for (int c = 0; c < 4; ++c) {
      const int col = C0 + c * 16 + nlo;
      const float bias = sB2[col];
      #pragma unroll
      for (int j = 0; j < 8; ++j)
        sOut[(Rb + rt * 16 + hb * 8 + j) * OSTR + col] = acc2[rt * 4 + c][j] + bias;
    }
  }
  __syncthreads();   // out2 rows were written by wave pairs

  // ---- reparameterize: z = mean[idx] + eps * exp(0.5*logvar[idx]) ----
  if (lane < 16) {
    const int b = wid * 16 + lane;
    const size_t gb = (size_t)g * batch + (size_t)tile * BT + b;
    const int idx = indices[gb];
    const float mean = sOut[b * OSTR + idx];
    const float lv   = sOut[b * OSTR + 64 + idx];
    out[gb] = mean + eps[gb] * __expf(0.5f * lv);
  }
}

extern "C" void kernel_launch(void* const* d_in, const int* in_sizes, int n_in,
                              void* d_out, int out_size, void* d_ws, size_t ws_size,
                              hipStream_t stream) {
  const float* X   = (const float*)d_in[0];
  const float* eps = (const float*)d_in[1];
  const float* W1  = (const float*)d_in[2];
  const float* b1  = (const float*)d_in[3];
  const float* W2  = (const float*)d_in[4];
  const float* b2  = (const float*)d_in[5];
  const int*   idx = (const int*)d_in[6];
  float* out = (float*)d_out;

  const int batch = in_sizes[1] / 8;   // eps is (8, BATCH)
  const int tiles = batch / BT;
  dim3 grid(8 * tiles), block(256);
  hipLaunchKernelGGL(envae_kernel, grid, block, 0, stream,
                     X, eps, W1, b1, W2, b2, idx, out, batch);
}
